// SoftmaxLoss_36438502539297
// MI455X (gfx1250) — matrix operand order repaired and checked by reading, not compiled, not run on hardware
//
#include <hip/hip_runtime.h>
#include <hip/hip_bf16.h>
#include <math.h>

// ---------------------------------------------------------------------------
// Softmax cross-entropy loss, fused bf16x2-split WMMA GEMM + online softmax.
// N=2048 rows, D=1024 features, C=128000 classes.
// ---------------------------------------------------------------------------

typedef __attribute__((ext_vector_type(16))) __bf16 v16bf;
typedef __attribute__((ext_vector_type(8)))  float  v8f;

static constexpr int N_     = 2048;
static constexpr int D_     = 1024;
static constexpr int C_     = 128000;
static constexpr int MROWS  = 32;            // rows per workgroup
static constexpr int CHUNKS = 50;            // class chunks (gridDim.y)
static constexpr int CPC    = C_ / CHUNKS;   // 2560 classes per chunk
static constexpr int CITER  = CPC / 128;     // fallback: 20 iters of 128 classes
static constexpr int CITER2 = CPC / 256;     // pre path: 10 iters of 256 classes

__device__ __forceinline__ unsigned short bf16_bits(__bf16 b) {
  return __builtin_bit_cast(unsigned short, b);
}

// A-fragment load from LDS: contiguous k order, same as B (lanes 0-15: K 0-15,
// lanes 16-31: K 16-31). Two contiguous 16-byte DS loads.
__device__ __forceinline__ v16bf load_a_frag(const unsigned short* base) {
  union { uint4 q[2]; v16bf v; } u;
  const uint4* p = (const uint4*)base;
  u.q[0] = p[0];   // K = kb .. kb+7
  u.q[1] = p[1];   // K = kb+8 .. kb+15
  return u.v;
}

// B-fragment: 32 contiguous bytes (16 bf16) -> v16bf
__device__ __forceinline__ v16bf make_b_frag(const uint4& q0, const uint4& q1) {
  union { uint4 q[2]; v16bf v; } u;
  u.q[0] = q0; u.q[1] = q1;
  return u.v;
}

// online softmax merge: (m,s) <- combine((m,s),(mo,so))
__device__ __forceinline__ void sm_merge(float& m, float& s, float mo, float so) {
  float mn = fmaxf(m, mo);
  s = s * __expf(m - mn) + so * __expf(mo - mn);
  m = mn;
}
// online softmax scalar update: absorb one logit x
__device__ __forceinline__ void sm_update(float& m, float& s, float x) {
  float mn = fmaxf(m, x);
  s = s * __expf(m - mn) + __expf(x - mn);
  m = mn;
}

// ---------------------------------------------------------------------------
// Kernel 0: one-shot weight fp32 -> bf16 hi/lo planes in workspace.
// ---------------------------------------------------------------------------
__global__ __launch_bounds__(256) void weight_cvt_kernel(
    const float* __restrict__ w, unsigned short* __restrict__ whi,
    unsigned short* __restrict__ wlo)
{
  size_t i = (size_t)blockIdx.x * 256 + threadIdx.x;    // float4 index
  float4 v = ((const float4*)w)[i];
  float xs[4] = {v.x, v.y, v.z, v.w};
  ushort4 h4, l4;
  unsigned short* hp = (unsigned short*)&h4;
  unsigned short* lp = (unsigned short*)&l4;
#pragma unroll
  for (int j = 0; j < 4; ++j) {
    __bf16 h = (__bf16)xs[j];
    hp[j] = bf16_bits(h);
    lp[j] = bf16_bits((__bf16)(xs[j] - (float)h));
  }
  ((ushort4*)whi)[i] = h4;
  ((ushort4*)wlo)[i] = l4;
}

// ---------------------------------------------------------------------------
// Shared epilogue: cross-lane + cross-wave (m,s) merge -> per-chunk partials.
// ---------------------------------------------------------------------------
__device__ __forceinline__ void partial_epilogue(
    float* rm, float* rs, float* redm, float* reds,
    int wave, int lane, int m16, int half, int rowBase,
    float* __restrict__ pm, float* __restrict__ ps)
{
#pragma unroll
  for (int i = 0; i < 16; ++i) {
    float m = rm[i], s = rs[i];
#pragma unroll
    for (int mask = 1; mask <= 8; mask <<= 1) {
      float mo = __shfl_xor(m, mask, 32);
      float so = __shfl_xor(s, mask, 32);
      sm_merge(m, s, mo, so);
    }
    rm[i] = m; rs[i] = s;
  }
  if (m16 == 0) {
#pragma unroll
    for (int i = 0; i < 16; ++i) {
      int mt = i >> 3, v = i & 7;
      int r = mt * 16 + half * 8 + v;
      redm[wave * MROWS + r] = rm[i];
      reds[wave * MROWS + r] = rs[i];
    }
  }
  __syncthreads();
  int tid = threadIdx.x;
  if (tid < MROWS) {
    float m = redm[tid], s = reds[tid];
#pragma unroll
    for (int w = 1; w < 8; ++w)
      sm_merge(m, s, redm[w * MROWS + tid], reds[w * MROWS + tid]);
    size_t idx = (size_t)blockIdx.y * N_ + rowBase + tid;
    pm[idx] = m;
    ps[idx] = s;
  }
}

// Stage features tile into LDS as bf16 hi/lo.
__device__ __forceinline__ void stage_a_tile(
    const float* __restrict__ feat, unsigned short* ahi, unsigned short* alo,
    int rowBase)
{
  const float4* f4 = (const float4*)feat;
  for (int i = threadIdx.x; i < MROWS * (D_ / 4); i += 256) {
    int r  = i >> 8;          // 256 float4 per row
    int c4 = i & 255;
    float4 v = f4[(size_t)(rowBase + r) * (D_ / 4) + c4];
    float xs[4] = {v.x, v.y, v.z, v.w};
#pragma unroll
    for (int j = 0; j < 4; ++j) {
      __bf16 h = (__bf16)xs[j];
      __bf16 l = (__bf16)(xs[j] - (float)h);
      ahi[r * D_ + c4 * 4 + j] = bf16_bits(h);
      alo[r * D_ + c4 * 4 + j] = bf16_bits(l);
    }
  }
  __syncthreads();
}

// ---------------------------------------------------------------------------
// Kernel 1a: GEMM partials from PRE-CONVERTED bf16 hi/lo weight planes.
// Each wave handles TWO 16-class B tiles per k-step: 8 ds_load_b128 feed
// 12 WMMAs (4 independent accumulator chains), halving LDS bytes per FLOP.
// ---------------------------------------------------------------------------
__global__ __launch_bounds__(256) void xent_partial_pre_kernel(
    const float* __restrict__ feat, const unsigned short* __restrict__ whi,
    const unsigned short* __restrict__ wlo, const float* __restrict__ bias,
    float* __restrict__ pm, float* __restrict__ ps)
{
  extern __shared__ char smem[];
  unsigned short* ahi = (unsigned short*)smem;       // [MROWS][D_]
  unsigned short* alo = ahi + MROWS * D_;            // [MROWS][D_]
  float* redm = (float*)(alo + MROWS * D_);          // [8][MROWS]
  float* reds = redm + 8 * MROWS;                    // [8][MROWS]

  const int tid    = threadIdx.x;
  const int wave   = tid >> 5;
  const int lane   = tid & 31;
  const int m16    = lane & 15;
  const int half   = lane >> 4;
  const int kb     = half * 16;
  const int rowBase = blockIdx.x * MROWS;

  stage_a_tile(feat, ahi, alo, rowBase);

  float rm[16], rs[16];
#pragma unroll
  for (int i = 0; i < 16; ++i) { rm[i] = -__builtin_inff(); rs[i] = 0.f; }

  const int chunkBase = blockIdx.y * CPC;

  for (int ci = 0; ci < CITER2; ++ci) {
    const int c0 = chunkBase + ci * 256 + wave * 32;   // two 16-class tiles
    // B 32x16 layout: lanes 0-15 hold K ks..ks+15, lanes 16-31 hold K ks+16..+31.
    const unsigned short* hrow0 = whi + (size_t)(c0 + m16)      * D_ + 16 * half;
    const unsigned short* lrow0 = wlo + (size_t)(c0 + m16)      * D_ + 16 * half;
    const unsigned short* hrow1 = whi + (size_t)(c0 + 16 + m16) * D_ + 16 * half;
    const unsigned short* lrow1 = wlo + (size_t)(c0 + 16 + m16) * D_ + 16 * half;

    v8f acc0a = {}; v8f acc1a = {};    // class tile 0, m-tiles 0/1
    v8f acc0b = {}; v8f acc1b = {};    // class tile 1, m-tiles 0/1

#pragma unroll 2
    for (int ks = 0; ks < D_; ks += 32) {
      // ---- B fragments: 8 global_load_b128 (clause-able) ----
      const uint4* ph0 = (const uint4*)(hrow0 + ks);
      const uint4* pl0 = (const uint4*)(lrow0 + ks);
      const uint4* ph1 = (const uint4*)(hrow1 + ks);
      const uint4* pl1 = (const uint4*)(lrow1 + ks);
      uint4 h00 = ph0[0], h01 = ph0[1];
      uint4 l00 = pl0[0], l01 = pl0[1];
      uint4 h10 = ph1[0], h11 = ph1[1];
      uint4 l10 = pl1[0], l11 = pl1[1];

      // ---- A fragments from LDS (two 16x16 row tiles, hi+lo) ----
      const int aoff = ks + kb;
      v16bf a0h = load_a_frag(ahi + (size_t)(0  + m16) * D_ + aoff);
      v16bf a0l = load_a_frag(alo + (size_t)(0  + m16) * D_ + aoff);
      v16bf a1h = load_a_frag(ahi + (size_t)(16 + m16) * D_ + aoff);
      v16bf a1l = load_a_frag(alo + (size_t)(16 + m16) * D_ + aoff);

      v16bf bh0 = make_b_frag(h00, h01);
      v16bf bl0 = make_b_frag(l00, l01);
      v16bf bh1 = make_b_frag(h10, h11);
      v16bf bl1 = make_b_frag(l10, l11);

      // ---- bf16x2-split: hi*hi + hi*lo + lo*hi for 4 output tiles ----
      acc0a = __builtin_amdgcn_wmma_f32_16x16x32_bf16(false, a0h, false, bh0, (short)0, acc0a, false, false);
      acc0b = __builtin_amdgcn_wmma_f32_16x16x32_bf16(false, a0h, false, bh1, (short)0, acc0b, false, false);
      acc1a = __builtin_amdgcn_wmma_f32_16x16x32_bf16(false, a1h, false, bh0, (short)0, acc1a, false, false);
      acc1b = __builtin_amdgcn_wmma_f32_16x16x32_bf16(false, a1h, false, bh1, (short)0, acc1b, false, false);
      acc0a = __builtin_amdgcn_wmma_f32_16x16x32_bf16(false, a0h, false, bl0, (short)0, acc0a, false, false);
      acc0b = __builtin_amdgcn_wmma_f32_16x16x32_bf16(false, a0h, false, bl1, (short)0, acc0b, false, false);
      acc1a = __builtin_amdgcn_wmma_f32_16x16x32_bf16(false, a1h, false, bl0, (short)0, acc1a, false, false);
      acc1b = __builtin_amdgcn_wmma_f32_16x16x32_bf16(false, a1h, false, bl1, (short)0, acc1b, false, false);
      acc0a = __builtin_amdgcn_wmma_f32_16x16x32_bf16(false, a0l, false, bh0, (short)0, acc0a, false, false);
      acc0b = __builtin_amdgcn_wmma_f32_16x16x32_bf16(false, a0l, false, bh1, (short)0, acc0b, false, false);
      acc1a = __builtin_amdgcn_wmma_f32_16x16x32_bf16(false, a1l, false, bh0, (short)0, acc1a, false, false);
      acc1b = __builtin_amdgcn_wmma_f32_16x16x32_bf16(false, a1l, false, bh1, (short)0, acc1b, false, false);
    }

    // ---- online softmax update; lane covers class columns m16 and 16+m16 ----
    const float bq0 = bias[c0 + m16];
    const float bq1 = bias[c0 + 16 + m16];
#pragma unroll
    for (int v = 0; v < 8; ++v) {
      sm_update(rm[v],     rs[v],     acc0a[v] + bq0);
      sm_update(rm[v],     rs[v],     acc0b[v] + bq1);
      sm_update(rm[8 + v], rs[8 + v], acc1a[v] + bq0);
      sm_update(rm[8 + v], rs[8 + v], acc1b[v] + bq1);
    }
  }

  partial_epilogue(rm, rs, redm, reds, wave, lane, m16, half, rowBase, pm, ps);
}

// ---------------------------------------------------------------------------
// Kernel 1b (fallback, small ws): stream fp32 weights, convert in-loop.
// ---------------------------------------------------------------------------
__global__ __launch_bounds__(256) void xent_partial_kernel(
    const float* __restrict__ feat, const float* __restrict__ weight,
    const float* __restrict__ bias, float* __restrict__ pm, float* __restrict__ ps)
{
  extern __shared__ char smem[];
  unsigned short* ahi = (unsigned short*)smem;
  unsigned short* alo = ahi + MROWS * D_;
  float* redm = (float*)(alo + MROWS * D_);
  float* reds = redm + 8 * MROWS;

  const int tid    = threadIdx.x;
  const int wave   = tid >> 5;
  const int lane   = tid & 31;
  const int m16    = lane & 15;
  const int half   = lane >> 4;
  const int kb     = half * 16;
  const int rowBase = blockIdx.x * MROWS;

  stage_a_tile(feat, ahi, alo, rowBase);

  float rm[16], rs[16];
#pragma unroll
  for (int i = 0; i < 16; ++i) { rm[i] = -__builtin_inff(); rs[i] = 0.f; }

  const int chunkBase = blockIdx.y * CPC;

  for (int ci = 0; ci < CITER; ++ci) {
    const int c0 = chunkBase + ci * 128 + wave * 16;
    const float* wrow = weight + (size_t)(c0 + m16) * D_;

    v8f acc0 = {}; v8f acc1 = {};

    float4 buf[2][4];
    {
      const float4* p = (const float4*)(wrow + 16 * half);
      buf[0][0] = p[0]; buf[0][1] = p[1]; buf[0][2] = p[2]; buf[0][3] = p[3];
    }

#pragma unroll 2
    for (int ks = 0; ks < D_; ks += 32) {
      const int cur = (ks >> 5) & 1;
      const int nxt = cur ^ 1;
      if (ks + 32 < D_) {
        const float4* pN = (const float4*)(wrow + (ks + 32) + 16 * half);
        buf[nxt][0] = pN[0]; buf[nxt][1] = pN[1];
        buf[nxt][2] = pN[2]; buf[nxt][3] = pN[3];
      }

      const int aoff = ks + kb;
      v16bf a0h = load_a_frag(ahi + (size_t)(0  + m16) * D_ + aoff);
      v16bf a0l = load_a_frag(alo + (size_t)(0  + m16) * D_ + aoff);
      v16bf a1h = load_a_frag(ahi + (size_t)(16 + m16) * D_ + aoff);
      v16bf a1l = load_a_frag(alo + (size_t)(16 + m16) * D_ + aoff);

      float4 q0 = buf[cur][0], q1 = buf[cur][1], q2 = buf[cur][2], q3 = buf[cur][3];
      float xs[16] = {q0.x,q0.y,q0.z,q0.w, q1.x,q1.y,q1.z,q1.w,
                      q2.x,q2.y,q2.z,q2.w, q3.x,q3.y,q3.z,q3.w};
      v16bf bhv, blv;
#pragma unroll
      for (int j = 0; j < 16; ++j) {
        __bf16 h = (__bf16)xs[j];
        bhv[j] = h;
        blv[j] = (__bf16)(xs[j] - (float)h);
      }

      acc0 = __builtin_amdgcn_wmma_f32_16x16x32_bf16(false, a0h, false, bhv, (short)0, acc0, false, false);
      acc0 = __builtin_amdgcn_wmma_f32_16x16x32_bf16(false, a0h, false, blv, (short)0, acc0, false, false);
      acc0 = __builtin_amdgcn_wmma_f32_16x16x32_bf16(false, a0l, false, bhv, (short)0, acc0, false, false);
      acc1 = __builtin_amdgcn_wmma_f32_16x16x32_bf16(false, a1h, false, bhv, (short)0, acc1, false, false);
      acc1 = __builtin_amdgcn_wmma_f32_16x16x32_bf16(false, a1h, false, blv, (short)0, acc1, false, false);
      acc1 = __builtin_amdgcn_wmma_f32_16x16x32_bf16(false, a1l, false, bhv, (short)0, acc1, false, false);
    }

    const float bq = bias[c0 + m16];
#pragma unroll
    for (int v = 0; v < 8; ++v) {
      sm_update(rm[v],     rs[v],     acc0[v] + bq);
      sm_update(rm[8 + v], rs[8 + v], acc1[v] + bq);
    }
  }

  partial_epilogue(rm, rs, redm, reds, wave, lane, m16, half, rowBase, pm, ps);
}

// ---------------------------------------------------------------------------
// Kernel 2: per-row loss (exact fp32 target dot + fixed-order partial merge).
// ---------------------------------------------------------------------------
__global__ __launch_bounds__(256) void xent_row_kernel(
    const float* __restrict__ feat, const float* __restrict__ weight,
    const float* __restrict__ bias, const int* __restrict__ target,
    const float* __restrict__ pm, const float* __restrict__ ps,
    float* __restrict__ lossArr)
{
  const int wave = threadIdx.x >> 5;
  const int lane = threadIdx.x & 31;
  const int row  = blockIdx.x * 8 + wave;
  const int tgt  = target[row];

  const float* fr = feat   + (size_t)row * D_;
  const float* wr = weight + (size_t)tgt * D_;
  float td = 0.f;
  for (int j = lane; j < D_; j += 32) td = fmaf(fr[j], wr[j], td);
#pragma unroll
  for (int mask = 16; mask >= 1; mask >>= 1) td += __shfl_xor(td, mask, 32);

  float m = -__builtin_inff(), s = 0.f;
  for (int ch = lane; ch < CHUNKS; ch += 32)
    sm_merge(m, s, pm[(size_t)ch * N_ + row], ps[(size_t)ch * N_ + row]);
#pragma unroll
  for (int mask = 16; mask >= 1; mask >>= 1) {
    float mo = __shfl_xor(m, mask, 32);
    float so = __shfl_xor(s, mask, 32);
    sm_merge(m, s, mo, so);
  }

  if (lane == 0) {
    float logit = td + bias[tgt];
    lossArr[row] = -(logit - m - __logf(s));
  }
}

// ---------------------------------------------------------------------------
// Kernel 3: fixed-order mean reduction to the scalar output.
// ---------------------------------------------------------------------------
__global__ __launch_bounds__(256) void xent_reduce_kernel(
    const float* __restrict__ lossArr, float* __restrict__ out)
{
  __shared__ float sm[256];
  float a = 0.f;
  for (int i = threadIdx.x; i < N_; i += 256) a += lossArr[i];
  sm[threadIdx.x] = a;
  __syncthreads();
  for (int st = 128; st >= 1; st >>= 1) {
    if (threadIdx.x < st) sm[threadIdx.x] += sm[threadIdx.x + st];
    __syncthreads();
  }
  if (threadIdx.x == 0) out[0] = sm[0] / (float)N_;
}

// ---------------------------------------------------------------------------
extern "C" void kernel_launch(void* const* d_in, const int* in_sizes, int n_in,
                              void* d_out, int out_size, void* d_ws, size_t ws_size,
                              hipStream_t stream) {
  const float* feat   = (const float*)d_in[0];   // [N, D] fp32
  const int*   target = (const int*)d_in[1];     // [N]
  const float* weight = (const float*)d_in[2];   // [C, D] fp32
  const float* bias   = (const float*)d_in[3];   // [C] fp32
  float* out = (float*)d_out;

  const size_t wElems      = (size_t)C_ * D_;
  const size_t wPlaneBytes = wElems * sizeof(unsigned short);          // 256 MB
  const size_t partialBytes = ((size_t)2 * CHUNKS * N_ + N_) * sizeof(float);
  const bool usePre = ws_size >= 2 * wPlaneBytes + partialBytes;

  const size_t smem = (size_t)MROWS * D_ * 2 * sizeof(unsigned short)
                    + (size_t)2 * 8 * MROWS * sizeof(float);
  dim3 g1(N_ / MROWS, CHUNKS);                   // 64 x 50 workgroups

  float* pm;
  if (usePre) {
    unsigned short* whi = (unsigned short*)d_ws;
    unsigned short* wlo = whi + wElems;
    pm = (float*)((char*)d_ws + 2 * wPlaneBytes);
    float* ps = pm + (size_t)CHUNKS * N_;

    weight_cvt_kernel<<<(unsigned)(wElems / 4 / 256), 256, 0, stream>>>(weight, whi, wlo);

    (void)hipFuncSetAttribute((const void*)xent_partial_pre_kernel,
                              hipFuncAttributeMaxDynamicSharedMemorySize, (int)smem);
    xent_partial_pre_kernel<<<g1, 256, smem, stream>>>(feat, whi, wlo, bias, pm, ps);
  } else {
    pm = (float*)d_ws;
    float* ps = pm + (size_t)CHUNKS * N_;
    (void)hipFuncSetAttribute((const void*)xent_partial_kernel,
                              hipFuncAttributeMaxDynamicSharedMemorySize, (int)smem);
    xent_partial_kernel<<<g1, 256, smem, stream>>>(feat, weight, bias, pm, ps);
  }

  float* ps      = pm + (size_t)CHUNKS * N_;
  float* lossArr = ps + (size_t)CHUNKS * N_;
  xent_row_kernel<<<N_ / 8, 256, 0, stream>>>(feat, weight, bias, target, pm, ps, lossArr);
  xent_reduce_kernel<<<1, 256, 0, stream>>>(lossArr, out);
}
